// RaftPlusDiclModule_25692494365422
// MI455X (gfx1250) — matrix operand order repaired and checked by reading, not compiled, not run on hardware
//
#include <hip/hip_runtime.h>
#include <hip/hip_bf16.h>

typedef _Float16 h16;
typedef __attribute__((ext_vector_type(16))) _Float16 v16h;
typedef __attribute__((ext_vector_type(8)))  _Float16 v8h;
typedef __attribute__((ext_vector_type(8)))  float    v8f;

#define NIMG 81
#define BN_EPS 1e-5f

__device__ __forceinline__ v8f wmma16(v16h a, v16h b, v8f c) {
  // D = A(16x32 f16) * B(32x16 f16) + C(16x16 f32)
  return __builtin_amdgcn_wmma_f32_16x16x32_f16(false, a, false, b, (short)0, c,
                                                false, false);
}
__device__ __forceinline__ v16h cat8(v8h lo, v8h hi) {
  return __builtin_shufflevector(lo, hi, 0, 1, 2, 3, 4, 5, 6, 7, 8, 9, 10, 11,
                                 12, 13, 14, 15);
}

// ---------------------------------------------------------------------------
// Zero the 1-pixel halo ring of a padded NHWC buffer [NIMG][Hp][Wp][C].
// ---------------------------------------------------------------------------
__global__ __launch_bounds__(256)
void zero_halo_kernel(h16* __restrict__ buf, int Hp, int Wp, int C) {
  const int img = blockIdx.y;
  const int i = blockIdx.x * 256 + threadIdx.x;
  const int perim = 2 * Wp + 2 * (Hp - 2);
  if (i >= perim * C) return;
  int pix = i / C, c = i - pix * C;
  int y, x;
  if (pix < Wp) { y = 0; x = pix; }
  else if (pix < 2 * Wp) { y = Hp - 1; x = pix - Wp; }
  else { int r = pix - 2 * Wp; y = 1 + (r >> 1); x = (r & 1) ? (Wp - 1) : 0; }
  buf[((long)(img * Hp + y) * Wp + x) * C + c] = (h16)0.f;
}

// ---------------------------------------------------------------------------
// Correlation feature builder. Padded NHWC out: x0[((d*66 + y+1)*66 + x+1)*64 + c]
// c<32: fmap1 (CHW f32), c>=32: bilinear sample of fmap2.
// ---------------------------------------------------------------------------
__global__ __launch_bounds__(256)
void corr_sample_kernel(h16* __restrict__ x0,
                        const float* __restrict__ f1,
                        const float* __restrict__ f2,
                        const float* __restrict__ coords,
                        int W2, int H2, float inv_scale) {
  const int d = blockIdx.y;                              // displacement 0..80
  const int p = blockIdx.x * blockDim.x + threadIdx.x;   // pixel 0..4095
  const int u = d / 9, v = d - u * 9;
  const float cenx = coords[p] * inv_scale + (float)(u - 4);
  const float ceny = coords[4096 + p] * inv_scale + (float)(v - 4);
  // normalize with fmap1 dims (63), sample at fmap2 dims [faithful quirk]
  const float xs = cenx * (float)(W2 - 1) * (1.0f / 63.0f);
  const float ys = ceny * (float)(H2 - 1) * (1.0f / 63.0f);
  const float xf = floorf(xs), yf = floorf(ys);
  const float wx = xs - xf, wy = ys - yf;
  const int ix0 = (int)xf, iy0 = (int)yf;
  const int ix1 = ix0 + 1, iy1 = iy0 + 1;
  const bool vx0 = (ix0 >= 0) & (ix0 < W2), vx1 = (ix1 >= 0) & (ix1 < W2);
  const bool vy0 = (iy0 >= 0) & (iy0 < H2), vy1 = (iy1 >= 0) & (iy1 < H2);
  const int cx0 = min(max(ix0, 0), W2 - 1), cx1 = min(max(ix1, 0), W2 - 1);
  const int cy0 = min(max(iy0, 0), H2 - 1), cy1 = min(max(iy1, 0), H2 - 1);
  const float w00 = (1.f - wx) * (1.f - wy) * (vx0 & vy0 ? 1.f : 0.f);
  const float w01 = wx * (1.f - wy) * (vx1 & vy0 ? 1.f : 0.f);
  const float w10 = (1.f - wx) * wy * (vx0 & vy1 ? 1.f : 0.f);
  const float w11 = wx * wy * (vx1 & vy1 ? 1.f : 0.f);
  const int hw2 = H2 * W2;

  h16 tmp[64];
#pragma unroll 8
  for (int c = 0; c < 32; ++c) tmp[c] = (h16)f1[(c << 12) + p];
#pragma unroll 8
  for (int c = 0; c < 32; ++c) {
    const float* fc = f2 + (long)c * hw2;
    float acc = w00 * fc[cy0 * W2 + cx0] + w01 * fc[cy0 * W2 + cx1] +
                w10 * fc[cy1 * W2 + cx0] + w11 * fc[cy1 * W2 + cx1];
    tmp[32 + c] = (h16)acc;
  }
  const int y = p >> 6, x = p & 63;
  v16h* dst = (v16h*)(x0 + ((long)(d * 66 + y + 1) * 66 + x + 1) * 64);
#pragma unroll
  for (int t = 0; t < 4; ++t) {
    v16h vv;
#pragma unroll
    for (int j = 0; j < 16; ++j) vv[j] = tmp[t * 16 + j];
    dst[t] = vv;
  }
}

// ---------------------------------------------------------------------------
// Implicit-GEMM 3x3 conv, padded NHWC in [Hi+2][Wi+2][CIN], tap-major K
// (k = tap*CIN + ci). Compile-time CIN => fully unrolled branch-free K loop.
// Each wave: 16 out-ch x 2 N-tiles (32 pixels) sharing the A fragment:
//   2x ds_load_b128 (A) + 4x global_load_b128 (B) + packed-f16 BN + 2x WMMA.
// Block 128 = 4 waves => 128 pixels per block.
// ---------------------------------------------------------------------------
template <int CIN, bool BN>
__global__ __launch_bounds__(128)
void conv3x3_wmma_kernel(h16* __restrict__ outh, float* __restrict__ outf,
                         const h16* __restrict__ in, const h16* __restrict__ A,
                         const float* __restrict__ ss,  // [CIN][2] or null
                         const float* __restrict__ bias,
                         int Cout, int Hi, int Wi, int Ho, int Wo, int stride) {
  constexpr int K = CIN * 9;
  __shared__ _Float16 Alds[16 * K];
  __shared__ h16 sscale[CIN];
  __shared__ h16 sshift[CIN];
  const int img = blockIdx.z;
  const int coBase = blockIdx.y << 4;
  const int tid = threadIdx.x;
  const int lane = tid & 31;
  const int wave = tid >> 5;
  const int n = lane & 15;   // pixel column within tile
  const int hi = lane >> 4;  // lane half
  const int pA = (blockIdx.x << 7) + (wave << 5) + n;
  const int pB = pA + 16;
  const int oyA = pA / Wo, oxA = pA - oyA * Wo;
  const int oyB = pB / Wo, oxB = pB - oyB * Wo;
  const int Wip = Wi + 2;
  const h16* inImg = in + (long)img * (Hi + 2) * Wip * CIN;

  if (BN && tid < CIN) {
    sscale[tid] = (h16)ss[2 * tid];
    sshift[tid] = (h16)ss[2 * tid + 1];
  }
  // stage full A tile (16 x K halves), vectorized
  for (int idx = tid * 8; idx < 16 * K; idx += 128 * 8) {
    int row = idx / K, col = idx - row * K;
    *(v8h*)&Alds[idx] = *(const v8h*)&A[(long)(coBase + row) * K + col];
  }
  __syncthreads();

  const int kb = hi ? 16 : 0;   // A half-lane offset
  const int kbb = hi ? 16 : 0;  // B half-lane offset
  const _Float16* arow = Alds + n * K + kb;
  const int yA = oyA * stride, xA = oxA * stride;  // padded coords
  const int yB = oyB * stride, xB = oxB * stride;
  const v16h zero16 = {};

  v8f accA = {}, accB = {};
#pragma unroll
  for (int tap = 0; tap < 9; ++tap) {
    const int ky = tap / 3, kx = tap - 3 * (tap / 3);
    const h16* bpA = inImg + ((long)(yA + ky) * Wip + (xA + kx)) * CIN + kbb;
    const h16* bpB = inImg + ((long)(yB + ky) * Wip + (xB + kx)) * CIN + kbb;
    const _Float16* ar = arow + tap * CIN;
#pragma unroll
    for (int cc = 0; cc < CIN; cc += 32) {
      v16h av = cat8(*(const v8h*)(ar + cc), *(const v8h*)(ar + cc + 8));
      v16h rawA = *(const v16h*)(bpA + cc);  // 32B aligned
      v16h rawB = *(const v16h*)(bpB + cc);
      v16h bvA, bvB;
      if (BN) {
        v16h scv = *(const v16h*)&sscale[cc + kbb];
        v16h sfv = *(const v16h*)&sshift[cc + kbb];
        bvA = __builtin_elementwise_max(rawA * scv + sfv, zero16);
        bvB = __builtin_elementwise_max(rawB * scv + sfv, zero16);
      } else {
        bvA = rawA;
        bvB = rawB;
      }
      accA = wmma16(av, bvA, accA);
      accB = wmma16(av, bvB, accB);
    }
  }

  if (outf) {  // f32 head output, unpadded [img][Ho*Wo]
    const float b0 = bias ? bias[0] : 0.0f;
    float* ob = outf + (long)img * Cout * (Ho * Wo);
#pragma unroll
    for (int r = 0; r < 8; ++r) {
      int co = coBase + r + (hi ? 8 : 0);
      if (co < Cout) {
        ob[(long)co * (Ho * Wo) + pA] = accA[r] + b0;
        ob[(long)co * (Ho * Wo) + pB] = accB[r] + b0;
      }
    }
  } else {  // fp16 NHWC padded interior
    v8h vA, vB;
#pragma unroll
    for (int r = 0; r < 8; ++r) { vA[r] = (h16)accA[r]; vB[r] = (h16)accB[r]; }
    long opixA = (long)(img * (Ho + 2) + oyA + 1) * (Wo + 2) + oxA + 1;
    long opixB = (long)(img * (Ho + 2) + oyB + 1) * (Wo + 2) + oxB + 1;
    *(v8h*)(outh + opixA * Cout + coBase + (hi ? 8 : 0)) = vA;
    *(v8h*)(outh + opixB * Cout + coBase + (hi ? 8 : 0)) = vB;
  }
}

// ---------------------------------------------------------------------------
// ConvTranspose2d 4x4 s2 p1 (64ch 32x32 -> 32ch 64x64) via WMMA.
// Padded NHWC in [34][34][64]; parity decomposition => branch-free taps.
// Ad[parity][32][256], k = tap*64 + ci. Out padded [66][66][32].
// Two N-tiles per wave sharing the A fragment.
// ---------------------------------------------------------------------------
__global__ __launch_bounds__(128)
void deconv_wmma_kernel(h16* __restrict__ out, const h16* __restrict__ in,
                        const h16* __restrict__ Ad, const float* __restrict__ ss) {
  __shared__ _Float16 Alds[16 * 256];
  __shared__ h16 sscale[64];
  __shared__ h16 sshift[64];
  const int img = blockIdx.z;
  const int coBase = (blockIdx.y >> 2) << 4;
  const int parity = blockIdx.y & 3;
  const int py = parity >> 1, px = parity & 1;
  const int tid = threadIdx.x, lane = tid & 31, wave = tid >> 5;
  const int n = lane & 15, hi = lane >> 4;
  const int qA = (blockIdx.x << 7) + (wave << 5) + n;  // parity-grid pixel
  const int qB = qA + 16;
  const int gyA = qA >> 5, gxA = qA & 31;
  const int gyB = qB >> 5, gxB = qB & 31;
  const int oyA = 2 * gyA + py, oxA = 2 * gxA + px;
  const int oyB = 2 * gyB + py, oxB = 2 * gxB + px;
  const h16* Ap = Ad + (long)parity * (32 * 256);
  const h16* inImg = in + (long)img * 34 * 34 * 64;

  if (tid < 64) { sscale[tid] = (h16)ss[2 * tid]; sshift[tid] = (h16)ss[2 * tid + 1]; }
  for (int idx = tid * 8; idx < 16 * 256; idx += 128 * 8) {
    int row = idx >> 8, col = idx & 255;
    *(v8h*)&Alds[idx] = *(const v8h*)&Ap[(coBase + row) * 256 + col];
  }
  __syncthreads();

  const int kb = hi ? 16 : 0;
  const int kbb = hi ? 16 : 0;
  const _Float16* arow = Alds + n * 256 + kb;
  const v16h zero16 = {};

  v8f accA = {}, accB = {};
#pragma unroll
  for (int tap = 0; tap < 4; ++tap) {
    const int ky = (1 - py) + ((tap >> 1) << 1);
    const int kx = (1 - px) + ((tap & 1) << 1);
    const int iyA = ((oyA + 1 - ky) >> 1) + 1;  // padded coords, in [0,33]
    const int ixA = ((oxA + 1 - kx) >> 1) + 1;
    const int iyB = ((oyB + 1 - ky) >> 1) + 1;
    const int ixB = ((oxB + 1 - kx) >> 1) + 1;
    const h16* bpA = inImg + ((long)iyA * 34 + ixA) * 64 + kbb;
    const h16* bpB = inImg + ((long)iyB * 34 + ixB) * 64 + kbb;
    const _Float16* ar = arow + tap * 64;
#pragma unroll
    for (int cc = 0; cc < 64; cc += 32) {
      v16h av = cat8(*(const v8h*)(ar + cc), *(const v8h*)(ar + cc + 8));
      v16h rawA = *(const v16h*)(bpA + cc);
      v16h rawB = *(const v16h*)(bpB + cc);
      v16h scv = *(const v16h*)&sscale[cc + kbb];
      v16h sfv = *(const v16h*)&sshift[cc + kbb];
      v16h bvA = __builtin_elementwise_max(rawA * scv + sfv, zero16);
      v16h bvB = __builtin_elementwise_max(rawB * scv + sfv, zero16);
      accA = wmma16(av, bvA, accA);
      accB = wmma16(av, bvB, accB);
    }
  }
  v8h vA, vB;
#pragma unroll
  for (int r = 0; r < 8; ++r) { vA[r] = (h16)accA[r]; vB[r] = (h16)accB[r]; }
  long opixA = (long)(img * 66 + oyA + 1) * 66 + oxA + 1;
  long opixB = (long)(img * 66 + oyB + 1) * 66 + oxB + 1;
  *(v8h*)(out + opixA * 32 + coBase + (hi ? 8 : 0)) = vA;
  *(v8h*)(out + opixB * 32 + coBase + (hi ? 8 : 0)) = vB;
}

// ---------------------------------------------------------------------------
// BN statistics over padded NHWC [NIMG][Hp][Wp][C] interior (H=W=2^lg).
// One block per 8-channel group; v8h strip loads; LDS tree-reduce; folds
// gamma/beta into per-channel {scale, shift}.
// ---------------------------------------------------------------------------
__global__ __launch_bounds__(256)
void bn_stats_kernel(const h16* __restrict__ x, const float* __restrict__ g,
                     const float* __restrict__ b, float* __restrict__ ss,
                     int C, int lg) {
  __shared__ float s1[256][8], s2[256][8];
  const int g8 = blockIdx.x * 8;
  const int tid = threadIdx.x;
  const int W = 1 << lg, Wp = W + 2;
  const int hwMask = (1 << (2 * lg)) - 1, xMask = W - 1;
  const long npix = (long)NIMG << (2 * lg);
  float a1[8], a2[8];
#pragma unroll
  for (int j = 0; j < 8; ++j) { a1[j] = 0.f; a2[j] = 0.f; }
  for (long p = tid; p < npix; p += 256) {
    int img = (int)(p >> (2 * lg));
    int rem = (int)p & hwMask;
    int y = rem >> lg, xx = rem & xMask;
    v8h vv = *(const v8h*)(x + ((long)(img * Wp + y + 1) * Wp + xx + 1) * C + g8);
#pragma unroll
    for (int j = 0; j < 8; ++j) {
      float v = (float)vv[j];
      a1[j] += v;
      a2[j] += v * v;
    }
  }
#pragma unroll
  for (int j = 0; j < 8; ++j) { s1[tid][j] = a1[j]; s2[tid][j] = a2[j]; }
  __syncthreads();
  for (int o = 128; o > 0; o >>= 1) {
    if (tid < o) {
#pragma unroll
      for (int j = 0; j < 8; ++j) {
        s1[tid][j] += s1[tid + o][j];
        s2[tid][j] += s2[tid + o][j];
      }
    }
    __syncthreads();
  }
  if (tid < 8) {
    int c = g8 + tid;
    float mean = s1[0][tid] / (float)npix;
    float var = s2[0][tid] / (float)npix - mean * mean;
    float sc = g[c] * rsqrtf(var + BN_EPS);
    ss[2 * c] = sc;
    ss[2 * c + 1] = b[c] - mean * sc;
  }
}

// ---------------------------------------------------------------------------
// DAP: out[o][p] = sum_d dap[o][d] * cost[d][p]
// ---------------------------------------------------------------------------
__global__ __launch_bounds__(256)
void dap_kernel(float* __restrict__ out, const float* __restrict__ cost,
                const float* __restrict__ dap) {
  const int o = blockIdx.y;
  const int p = blockIdx.x * blockDim.x + threadIdx.x;
  float acc = 0.f;
#pragma unroll 3
  for (int d = 0; d < 81; ++d)
    acc = fmaf(dap[o * 81 + d], cost[((long)d << 12) + p], acc);
  out[((long)o << 12) + p] = acc;
}

// ---------------------------------------------------------------------------
// Weight packing: tap-major K layout  A[co][tap*Cin + ci] = w[co][ci][ky][kx]
// ---------------------------------------------------------------------------
__global__ void pack_w3x3_kernel(h16* __restrict__ dst,
                                 const float* __restrict__ src, int Cout,
                                 int Cin) {
  const int K = Cin * 9;
  int i = blockIdx.x * blockDim.x + threadIdx.x;
  if (i >= Cout * K) return;
  int co = i / K, r = i - co * K;
  int tap = r / Cin, ci = r - tap * Cin;
  dst[i] = (h16)src[(co * Cin + ci) * 9 + tap];
}
__global__ void pack_wo_kernel(h16* __restrict__ dst,
                               const float* __restrict__ wo) {
  int i = blockIdx.x * blockDim.x + threadIdx.x;  // 16 * 288
  if (i >= 16 * 288) return;
  int co = i / 288, r = i - co * 288;
  int tap = r >> 5, ci = r & 31;
  dst[i] = (h16)(co == 0 ? wo[ci * 9 + tap] : 0.0f);
}
__global__ void pack_wd_kernel(h16* __restrict__ dst,
                               const float* __restrict__ wd) {
  int i = blockIdx.x * blockDim.x + threadIdx.x;  // 4*32*256
  if (i >= 4 * 32 * 256) return;
  int parity = i >> 13, rem = i & 8191;
  int co = rem >> 8, r = rem & 255;
  int tap = r >> 6, ci = r & 63;
  int py = parity >> 1, px = parity & 1;
  int ky = (1 - py) + ((tap >> 1) << 1);
  int kx = (1 - px) + ((tap & 1) << 1);
  dst[i] = (h16)wd[((ci * 32 + co) * 4 + ky) * 4 + kx];
}

// ---------------------------------------------------------------------------
extern "C" void kernel_launch(void* const* d_in, const int* in_sizes, int n_in,
                              void* d_out, int out_size, void* d_ws, size_t ws_size,
                              hipStream_t stream) {
  (void)in_sizes; (void)n_in; (void)out_size; (void)ws_size;
  char* ws = (char*)d_ws;
  size_t off = 0;
  auto take = [&](size_t bytes) {
    void* p = ws + off;
    off = (off + bytes + 255) & ~size_t(255);
    return p;
  };
  // padded NHWC activations
  h16*  x0 = (h16*)take(81ull * 66 * 66 * 64 * 2);
  h16*  y1 = (h16*)take(81ull * 66 * 66 * 96 * 2);
  h16*  y2 = (h16*)take(81ull * 34 * 34 * 128 * 2);
  h16*  y3 = (h16*)take(81ull * 34 * 34 * 128 * 2);
  h16*  y4 = (h16*)take(81ull * 34 * 34 * 64 * 2);
  h16*  y5 = (h16*)take(81ull * 66 * 66 * 32 * 2);
  float* cost = (float*)take(81ull * 4096 * 4);  // [d][p]
  h16*  A1 = (h16*)take(96ull * 576 * 2);
  h16*  A2 = (h16*)take(128ull * 864 * 2);
  h16*  A3 = (h16*)take(128ull * 1152 * 2);
  h16*  A4 = (h16*)take(64ull * 1152 * 2);
  h16*  Ad = (h16*)take(4ull * 32 * 256 * 2);
  h16*  Ao = (h16*)take(16ull * 288 * 2);
  float* ss1 = (float*)take(128 * 2 * 4);
  float* ss2 = (float*)take(128 * 2 * 4);
  float* ss3 = (float*)take(128 * 2 * 4);
  float* ss4 = (float*)take(128 * 2 * 4);
  float* ss5 = (float*)take(128 * 2 * 4);

  // zero halos once per call (interiors are fully overwritten each level)
  auto zhalo = [&](h16* buf, int Hp, int Wp, int C) {
    int nel = (2 * Wp + 2 * (Hp - 2)) * C;
    zero_halo_kernel<<<dim3((nel + 255) / 256, NIMG), 256, 0, stream>>>(buf, Hp,
                                                                        Wp, C);
  };
  zhalo(x0, 66, 66, 64);
  zhalo(y1, 66, 66, 96);
  zhalo(y2, 34, 34, 128);
  zhalo(y3, 34, 34, 128);
  zhalo(y4, 34, 34, 64);
  zhalo(y5, 66, 66, 32);

  const float* coords = (const float*)d_in[8];

  for (int lvl = 0; lvl < 4; ++lvl) {
    const float* f1 = (const float*)d_in[2 * lvl];
    const float* f2 = (const float*)d_in[2 * lvl + 1];
    void* const* P = (void* const*)(d_in + 9 + lvl * 18);  // jax-sorted leaves
    const float* b1  = (const float*)P[0];
    const float* b2  = (const float*)P[1];
    const float* b3  = (const float*)P[2];
    const float* b4  = (const float*)P[3];
    const float* b5  = (const float*)P[4];
    const float* bo  = (const float*)P[5];
    const float* dap = (const float*)P[6];
    const float* g1  = (const float*)P[7];
    const float* g2  = (const float*)P[8];
    const float* g3  = (const float*)P[9];
    const float* g4  = (const float*)P[10];
    const float* g5  = (const float*)P[11];
    const float* w1  = (const float*)P[12];
    const float* w2  = (const float*)P[13];
    const float* w3  = (const float*)P[14];
    const float* w4  = (const float*)P[15];
    const float* wd  = (const float*)P[16];
    const float* wo  = (const float*)P[17];

    auto pack = [&](h16* dst, const float* src, int Cout, int Cin) {
      int nel = Cout * Cin * 9;
      pack_w3x3_kernel<<<(nel + 255) / 256, 256, 0, stream>>>(dst, src, Cout,
                                                              Cin);
    };
    pack(A1, w1, 96, 64);
    pack(A2, w2, 128, 96);
    pack(A3, w3, 128, 128);
    pack(A4, w4, 64, 128);
    pack_wd_kernel<<<(4 * 32 * 256 + 255) / 256, 256, 0, stream>>>(Ad, wd);
    pack_wo_kernel<<<(16 * 288 + 255) / 256, 256, 0, stream>>>(Ao, wo);

    int W2 = 64 >> lvl, H2 = 64 >> lvl;
    corr_sample_kernel<<<dim3(16, 81), 256, 0, stream>>>(
        x0, f1, f2, coords, W2, H2, 1.0f / (float)(1 << lvl));

    // conv1: 64 -> 96, 64x64
    conv3x3_wmma_kernel<64, false><<<dim3(32, 6, 81), 128, 0, stream>>>(
        y1, nullptr, x0, A1, nullptr, nullptr, 96, 64, 64, 64, 64, 1);
    bn_stats_kernel<<<12, 256, 0, stream>>>(y1, g1, b1, ss1, 96, 6);
    // conv2: 96 -> 128, stride 2 -> 32x32
    conv3x3_wmma_kernel<96, true><<<dim3(8, 8, 81), 128, 0, stream>>>(
        y2, nullptr, y1, A2, ss1, nullptr, 128, 64, 64, 32, 32, 2);
    bn_stats_kernel<<<16, 256, 0, stream>>>(y2, g2, b2, ss2, 128, 5);
    // conv3: 128 -> 128, 32x32
    conv3x3_wmma_kernel<128, true><<<dim3(8, 8, 81), 128, 0, stream>>>(
        y3, nullptr, y2, A3, ss2, nullptr, 128, 32, 32, 32, 32, 1);
    bn_stats_kernel<<<16, 256, 0, stream>>>(y3, g3, b3, ss3, 128, 5);
    // conv4: 128 -> 64, 32x32
    conv3x3_wmma_kernel<128, true><<<dim3(8, 4, 81), 128, 0, stream>>>(
        y4, nullptr, y3, A4, ss3, nullptr, 64, 32, 32, 32, 32, 1);
    bn_stats_kernel<<<8, 256, 0, stream>>>(y4, g4, b4, ss4, 64, 5);
    // deconv: 64 -> 32, 32x32 -> 64x64 (parity-decomposed)
    deconv_wmma_kernel<<<dim3(8, 8, 81), 128, 0, stream>>>(y5, y4, Ad, ss4);
    bn_stats_kernel<<<4, 256, 0, stream>>>(y5, g5, b5, ss5, 32, 6);
    // head: 32 -> 1 + bias, f32 out
    conv3x3_wmma_kernel<32, true><<<dim3(32, 1, 81), 128, 0, stream>>>(
        nullptr, cost, y5, Ao, ss5, bo, 1, 64, 64, 64, 64, 1);

    float* out_lvl = (float*)d_out + (long)lvl * 81 * 4096;
    dap_kernel<<<dim3(16, 81), 256, 0, stream>>>(out_lvl, cost, dap);
  }
}
